// UpSample_7945689498136
// MI455X (gfx1250) — matrix operand order repaired and checked by reading, not compiled, not run on hardware
//
#include <hip/hip_runtime.h>
#include <hip/hip_bf16.h>
#include <float.h>

#define BS 16
#define N_STK 256
#define N_BEF 512
#define N_PNT 64
#define EMB 256
#define SP_IN 256
#define SP_OUT 128
#define DN_IN 128
#define DN_OUT 64
#define KNN_K 3
#define BN_EPS 1e-5f
#define NEG_SLOPE 0.2f

typedef _Float16 half_t;
typedef __attribute__((ext_vector_type(16))) half_t v16h;
typedef __attribute__((ext_vector_type(8)))  half_t v8h;
typedef __attribute__((ext_vector_type(8)))  float  v8f;

__device__ __forceinline__ v16h concat8(v8h lo, v8h hi) {
    return __builtin_shufflevector(lo, hi, 0,1,2,3,4,5,6,7,8,9,10,11,12,13,14,15);
}

// A operand (16x32 f16, M x K): per-lane row M = lane&15.
// lanes 0-15 hold K = kbase+0..15; lanes 16-31 hold K = kbase+16..31.
__device__ __forceinline__ v16h load_a_row(const half_t* Wrow, int kbase, bool laneHi) {
    const int s0 = laneHi ? 16 : 0;
    const int s1 = laneHi ? 24 : 8;
    v8h lo = *(const v8h*)(Wrow + kbase + s0);
    v8h hi = *(const v8h*)(Wrow + kbase + s1);
    return concat8(lo, hi);
}

// B operand (32x16 f16, K x N): per-lane col N = lane&15, X stored transposed
// (one column's K values contiguous). lanes 0-15: K=kbase+0..15; lanes 16-31: +16..31.
__device__ __forceinline__ v16h load_b_col(const half_t* Xrow, int kbase, bool laneHi) {
    const half_t* p = Xrow + kbase + (laneHi ? 16 : 0);
    v8h lo = *(const v8h*)(p);
    v8h hi = *(const v8h*)(p + 8);
    return concat8(lo, hi);
}

// ---------------------------------------------------------------------------
// Kernel 1: repack weights to f16 and fold BN+bias into per-channel (A, B)
// ---------------------------------------------------------------------------
__global__ void repack_kernel(const float* __restrict__ ct_w, const float* __restrict__ ct_b,
                              const float* __restrict__ bn_g, const float* __restrict__ bn_b,
                              const float* __restrict__ bn_mean, const float* __restrict__ bn_var,
                              const float* __restrict__ sp_w, const float* __restrict__ sp_b,
                              const float* __restrict__ sp_g, const float* __restrict__ sp_beta,
                              const float* __restrict__ sp_mean, const float* __restrict__ sp_var,
                              half_t* __restrict__ We, half_t* __restrict__ Wo,
                              half_t* __restrict__ spwh,
                              float* __restrict__ As, float* __restrict__ Bs,
                              float* __restrict__ Ad, float* __restrict__ Bd) {
    const int i = blockIdx.x * blockDim.x + threadIdx.x;
    if (i < SP_OUT * SP_IN) spwh[i] = (half_t)sp_w[i];
    if (i < DN_OUT * 2 * DN_IN) {           // 64 * 256
        const int oc = i >> 8;
        const int kk = i & 255;
        const int ic = kk & 127;
        const bool hi = kk >= 128;
        // torch ConvTranspose2d weight layout (in=128, out=64, 1, 4)
        We[i] = (half_t)ct_w[ic * 256 + oc * 4 + (hi ? 3 : 1)];  // even outputs: taps 1,3
        Wo[i] = (half_t)ct_w[ic * 256 + oc * 4 + (hi ? 2 : 0)];  // odd outputs:  taps 0,2
    }
    if (i < SP_OUT) {
        const float a = sp_g[i] * rsqrtf(sp_var[i] + BN_EPS);
        As[i] = a;
        Bs[i] = sp_beta[i] + a * (sp_b[i] - sp_mean[i]);
    }
    if (i < DN_OUT) {
        const float a = bn_g[i] * rsqrtf(bn_var[i] + BN_EPS);
        Ad[i] = a;
        Bd[i] = bn_b[i] + a * (ct_b[i] - bn_mean[i]);
    }
}

// ---------------------------------------------------------------------------
// Kernel 2: 3-NN per (batch, before-stroke) + inverse-distance weights
// ---------------------------------------------------------------------------
__global__ void knn_kernel(const float* __restrict__ stk,   // [bs][EMB][N_STK]
                           const float* __restrict__ bef,   // [bs][EMB][N_BEF]
                           int* __restrict__ idx_ws, float* __restrict__ w_ws) {
    const int b = blockIdx.x >> 9;
    const int s = blockIdx.x & 511;
    const int tid = threadIdx.x;   // 256
    __shared__ float befv[EMB];
    __shared__ float d2s[N_STK];
    befv[tid] = bef[(b * EMB + tid) * N_BEF + s];
    __syncthreads();
    const float* col = stk + b * EMB * N_STK + tid;   // stk[b][e][tid]
    float acc = 0.f;
    #pragma unroll 4
    for (int e = 0; e < EMB; ++e) {
        const float d = befv[e] - col[e * N_STK];
        acc = fmaf(d, d, acc);
    }
    d2s[tid] = acc;
    __syncthreads();
    if (tid == 0) {
        int bi[KNN_K]; float bd[KNN_K];
        for (int j = 0; j < KNN_K; ++j) {
            float best = FLT_MAX; int bx = 0;
            for (int t = 0; t < N_STK; ++t) {
                const float v = d2s[t];
                if (v < best) { best = v; bx = t; }
            }
            bi[j] = bx; bd[j] = best; d2s[bx] = FLT_MAX;
        }
        float r[KNN_K], sum = 0.f;
        for (int j = 0; j < KNN_K; ++j) { r[j] = 1.0f / (bd[j] + 1e-8f); sum += r[j]; }
        const int base = (b * N_BEF + s) * KNN_K;
        for (int j = 0; j < KNN_K; ++j) { idx_ws[base + j] = bi[j]; w_ws[base + j] = r[j] / sum; }
    }
}

// ---------------------------------------------------------------------------
// Kernel 3: sparse branch — gather-interp into LDS, WMMA GEMM 128x256 * 256x32
// One workgroup per (batch, 32 output columns). 256 thr = 8 waves.
// ---------------------------------------------------------------------------
__global__ void sparse_kernel(const float* __restrict__ sparse_fea,   // [bs][256][256]
                              const int* __restrict__ idx_ws, const float* __restrict__ w_ws,
                              const half_t* __restrict__ spwh,        // [128][256] f16
                              const float* __restrict__ As, const float* __restrict__ Bs,
                              float* __restrict__ out_sp) {            // [bs][128][512]
    const int b  = blockIdx.x >> 4;
    const int s0 = (blockIdx.x & 15) * 32;
    const int tid = threadIdx.x;
    __shared__ half_t Xs[32][SP_IN];   // transposed: column n -> contiguous K (16 KB)
    __shared__ int   i3[32][KNN_K];
    __shared__ float w3[32][KNN_K];
    if (tid < 32 * KNN_K) {
        const int n = tid / KNN_K, j = tid % KNN_K;
        const int base = (b * N_BEF + s0 + n) * KNN_K + j;
        i3[n][j] = idx_ws[base];
        w3[n][j] = w_ws[base];
    }
    __syncthreads();
    for (int t = tid; t < 32 * SP_IN; t += 256) {
        const int n = t >> 8, c = t & 255;
        const float* row = sparse_fea + (b * SP_IN + c) * N_STK;
        const float v = w3[n][0] * row[i3[n][0]] + w3[n][1] * row[i3[n][1]] + w3[n][2] * row[i3[n][2]];
        Xs[n][c] = (half_t)v;
    }
    __syncthreads();
    const int wv = tid >> 5, lane = tid & 31, ln = lane & 15;
    const bool laneHi = lane >= 16;
    for (int task = wv; task < 16; task += 8) {          // 8 m-tiles x 2 n-tiles
        const int mt = task >> 1, nt = task & 1;
        const half_t* Wrow = spwh + (mt * 16 + ln) * SP_IN;
        const half_t* Xrow = &Xs[nt * 16 + ln][0];
        v8f c = {};
        #pragma unroll
        for (int kb = 0; kb < SP_IN; kb += 32) {
            const v16h a  = load_a_row(Wrow, kb, laneHi);
            const v16h bb = load_b_col(Xrow, kb, laneHi);
            c = __builtin_amdgcn_wmma_f32_16x16x32_f16(false, a, false, bb, (short)0, c, false, false);
        }
        const int col = s0 + nt * 16 + ln;
        #pragma unroll
        for (int r = 0; r < 8; ++r) {
            const int o = mt * 16 + r + (laneHi ? 8 : 0);
            float v = As[o] * c[r] + Bs[o];
            v = v >= 0.f ? v : NEG_SLOPE * v;
            out_sp[(b * SP_OUT + o) * N_BEF + col] = v;
        }
    }
}

// ---------------------------------------------------------------------------
// Kernel 4: dense branch — gather-interp tile into LDS (zero-padded, transposed),
// ConvTranspose as two phase-GEMMs with augmented K=256 via WMMA, fused BN+leaky.
// One workgroup per (batch, before-stroke). 256 thr = 8 waves.
// ---------------------------------------------------------------------------
__global__ void dense_kernel(const float* __restrict__ dense_fea,   // [bs][128][16384]
                             const int* __restrict__ idx_ws, const float* __restrict__ w_ws,
                             const half_t* __restrict__ We, const half_t* __restrict__ Wo, // [64][256]
                             const float* __restrict__ Ad, const float* __restrict__ Bd,
                             float* __restrict__ out2) {              // [bs][64][512][128]
    const int b = blockIdx.x >> 9;
    const int s = blockIdx.x & 511;
    const int tid = threadIdx.x;
    __shared__ half_t Xp[66][DN_IN];   // row j = point iw+1 (rows 0,65 zero), 16.5 KB
    __shared__ int   i3[KNN_K];
    __shared__ float w3[KNN_K];
    if (tid < KNN_K) {
        const int base = (b * N_BEF + s) * KNN_K + tid;
        i3[tid] = idx_ws[base];
        w3[tid] = w_ws[base];
    }
    if (tid < DN_IN) { Xp[0][tid] = (half_t)0.f; Xp[65][tid] = (half_t)0.f; }
    __syncthreads();
    for (int t = tid; t < DN_IN * N_PNT; t += 256) {
        const int ic = t >> 6, iw = t & 63;
        const float* row = dense_fea + (b * DN_IN + ic) * (N_STK * N_PNT);
        const float v = w3[0] * row[i3[0] * N_PNT + iw]
                      + w3[1] * row[i3[1] * N_PNT + iw]
                      + w3[2] * row[i3[2] * N_PNT + iw];
        Xp[iw + 1][ic] = (half_t)v;
    }
    __syncthreads();
    const int wv = tid >> 5, lane = tid & 31, ln = lane & 15;
    const bool laneHi = lane >= 16;
    // 2 phases x 4 m-tiles x 4 n-tiles = 32 tile tasks, 4 per wave
    for (int task = wv; task < 32; task += 8) {
        const int phase = task >> 4;            // 0 = even outputs, 1 = odd
        const int mt = (task >> 2) & 3;
        const int nt = task & 3;
        const half_t* Wph  = phase ? Wo : We;
        const half_t* Wrow = Wph + (mt * 16 + ln) * 256;
        const int m  = nt * 16 + ln;            // point index 0..63
        const int jA = phase ? 2 : 1;           // first K-half shift (rows of Xp)
        const int jB = phase ? 1 : 0;           // second K-half shift
        v8f c = {};
        #pragma unroll
        for (int kb = 0; kb < 256; kb += 32) {
            const bool second = kb >= 128;
            const half_t* Xrow = &Xp[m + (second ? jB : jA)][0];
            const int kloc = second ? (kb - 128) : kb;
            const v16h a  = load_a_row(Wrow, kb, laneHi);
            const v16h bb = load_b_col(Xrow, kloc, laneHi);
            c = __builtin_amdgcn_wmma_f32_16x16x32_f16(false, a, false, bb, (short)0, c, false, false);
        }
        const int ow = 2 * m + phase;
        #pragma unroll
        for (int r = 0; r < 8; ++r) {
            const int oc = mt * 16 + r + (laneHi ? 8 : 0);
            float v = Ad[oc] * c[r] + Bd[oc];
            v = v >= 0.f ? v : NEG_SLOPE * v;
            out2[((b * DN_OUT + oc) * N_BEF + s) * 128 + ow] = v;
        }
    }
}

// ---------------------------------------------------------------------------
extern "C" void kernel_launch(void* const* d_in, const int* in_sizes, int n_in,
                              void* d_out, int out_size, void* d_ws, size_t ws_size,
                              hipStream_t stream) {
    const float* sparse_fea = (const float*)d_in[0];
    const float* dense_fea  = (const float*)d_in[1];
    const float* stk_fea    = (const float*)d_in[2];
    const float* stk_bef    = (const float*)d_in[3];
    const float* sp_w    = (const float*)d_in[4];
    const float* sp_b    = (const float*)d_in[5];
    const float* sp_g    = (const float*)d_in[6];
    const float* sp_beta = (const float*)d_in[7];
    const float* sp_mean = (const float*)d_in[8];
    const float* sp_var  = (const float*)d_in[9];
    const float* ct_w    = (const float*)d_in[10];
    const float* ct_b    = (const float*)d_in[11];
    const float* bn_g    = (const float*)d_in[12];
    const float* bn_b    = (const float*)d_in[13];
    const float* bn_mean = (const float*)d_in[14];
    const float* bn_var  = (const float*)d_in[15];

    // workspace layout (all sections >= 256B aligned)
    char* ws = (char*)d_ws;
    int*    idx_ws = (int*)   (ws);               // 16*512*3 ints  = 98304 B
    float*  w_ws   = (float*) (ws + 98304);       // 16*512*3 f32   = 98304 B
    half_t* We     = (half_t*)(ws + 196608);      // 64*256 f16     = 32768 B
    half_t* Wo     = (half_t*)(ws + 229376);      // 64*256 f16     = 32768 B
    half_t* spwh   = (half_t*)(ws + 262144);      // 128*256 f16    = 65536 B
    float*  As     = (float*) (ws + 327680);
    float*  Bs     = (float*) (ws + 328192);
    float*  Ad     = (float*) (ws + 328704);
    float*  Bd     = (float*) (ws + 328960);

    float* out_sp = (float*)d_out;                        // [16][128][512]
    float* out2   = (float*)d_out + BS * SP_OUT * N_BEF;  // [16][64][512][128]

    repack_kernel<<<128, 256, 0, stream>>>(ct_w, ct_b, bn_g, bn_b, bn_mean, bn_var,
                                           sp_w, sp_b, sp_g, sp_beta, sp_mean, sp_var,
                                           We, Wo, spwh, As, Bs, Ad, Bd);
    knn_kernel<<<BS * N_BEF, 256, 0, stream>>>(stk_fea, stk_bef, idx_ws, w_ws);
    sparse_kernel<<<BS * 16, 256, 0, stream>>>(sparse_fea, idx_ws, w_ws, spwh, As, Bs, out_sp);
    dense_kernel<<<BS * N_BEF, 256, 0, stream>>>(dense_fea, idx_ws, w_ws, We, Wo, Ad, Bd, out2);
}
